// ModulatedDeformConv_4389456576586
// MI455X (gfx1250) — matrix operand order repaired and checked by reading, not compiled, not run on hardware
//
#include <hip/hip_runtime.h>
#include <hip/hip_bf16.h>

typedef __attribute__((ext_vector_type(16))) __bf16 v16bf;
typedef __attribute__((ext_vector_type(8)))  __bf16 v8bf;
typedef __attribute__((ext_vector_type(8)))  float  v8f;

// Problem constants (from reference): N=8, Cin=Cout=256, H=W=56, 3x3, s=1,p=1,d=1
constexpr int Hc   = 56;
constexpr int Wc   = 56;
constexpr int HWc  = Hc * Wc;      // 3136
constexpr int CIN  = 256;
constexpr int COUT = 256;
constexpr int TAPS = 9;
constexpr int Nb   = 8;

constexpr int TP   = 64;           // pixels per workgroup tile (3136 % 64 == 0)
constexpr int KC   = 32;           // channel slice per WMMA K-step (bf16 K=32)
constexpr int WSTR = 40;           // LDS row stride (bf16) for weight tile  (32 + pad)
constexpr int VSTR = 40;           // LDS row stride (bf16) for value tile   (32 + pad)

static __device__ __forceinline__ __bf16 f2bf(float f) {
    // round-to-nearest-even fp32 -> bf16
    unsigned u = __builtin_bit_cast(unsigned, f);
    unsigned r = (u + 0x7FFFu + ((u >> 16) & 1u)) >> 16;
    return __builtin_bit_cast(__bf16, (unsigned short)r);
}

__global__ __launch_bounds__(256, 1)
void mdcn_wmma_kernel(const float* __restrict__ x,
                      const float* __restrict__ offs,
                      const float* __restrict__ msk,
                      const float* __restrict__ wgt,
                      const float* __restrict__ bias,
                      float* __restrict__ out) {
    const int tid  = threadIdx.x;
    const int lane = tid & 31;
    const int wave = tid >> 5;          // 8 waves: each owns 32 Cout rows
    const int half = lane >> 4;         // hi/lo 16-lane group (WMMA layout)
    const int m16  = lane & 15;

    const int n   = blockIdx.x / (HWc / TP);      // image
    const int hw0 = (blockIdx.x % (HWc / TP)) * TP;

    // LDS: weight tile [256 o][32 c], value tile [64 p][32 c], bilinear setup
    __shared__ __align__(16) __bf16 Wlds[COUT * WSTR];   // 20480 B
    __shared__ __align__(16) __bf16 Vlds[TP   * VSTR];   //  5120 B
    __shared__ int   Poff[4 * TP];                       // clamped corner flat offsets
    __shared__ float Pwt [4 * TP];                       // bilinear*mask*valid weights

    v8f acc[2][4];
#pragma unroll
    for (int t = 0; t < 2; ++t)
#pragma unroll
        for (int u = 0; u < 4; ++u)
            acc[t][u] = v8f{};

    for (int k = 0; k < TAPS; ++k) {
        // ---- per-tap bilinear setup: one thread per pixel ----
        if (tid < TP) {
            const int p  = tid;
            const int hw = hw0 + p;
            const int h  = hw / Wc;
            const int w  = hw % Wc;
            const int ob = (n * TAPS + k);
            const float dy = offs[(ob * 2    ) * HWc + hw];
            const float dx = offs[(ob * 2 + 1) * HWc + hw];
            const float mm = msk [ ob          * HWc + hw];
            const float py = dy + (float)(k / 3) + (float)(h - 1);
            const float px = dx + (float)(k % 3) + (float)(w - 1);
            const float fy = floorf(py), fx = floorf(px);
            const int y0 = (int)fy, x0 = (int)fx;
            const float wy1 = py - fy, wx1 = px - fx;
            const float wy0 = 1.0f - wy1, wx0 = 1.0f - wx1;
#pragma unroll
            for (int c2 = 0; c2 < 4; ++c2) {
                const int yy = y0 + (c2 >> 1);
                const int xx = x0 + (c2 & 1);
                const bool valid = (yy >= 0) & (yy < Hc) & (xx >= 0) & (xx < Wc);
                const float wb = ((c2 >> 1) ? wy1 : wy0) * ((c2 & 1) ? wx1 : wx0);
                const int yc = min(max(yy, 0), Hc - 1);
                const int xc = min(max(xx, 0), Wc - 1);
                Poff[c2 * TP + p] = yc * Wc + xc;
                Pwt [c2 * TP + p] = valid ? wb * mm : 0.0f;
            }
        }
        __syncthreads();

        for (int cb = 0; cb < CIN / KC; ++cb) {
            // ---- stage weights: thread t fills Cout row t (32 ch, fp32->bf16) ----
            {
                const int o = tid;
                const float* wsrc = wgt + (o * CIN + cb * KC) * TAPS + k;
                __bf16* dst = &Wlds[o * WSTR];
#pragma unroll
                for (int q = 0; q < 4; ++q) {
                    v8bf pk;
#pragma unroll
                    for (int j = 0; j < 8; ++j)
                        pk[j] = f2bf(wsrc[(q * 8 + j) * TAPS]);
                    *(v8bf*)(dst + q * 8) = pk;
                }
            }
            // ---- gather: thread -> (pixel p = tid/4, 8 channels) ----
            {
                const int p  = tid >> 2;
                const int c0 = (tid & 3) * 8;
                const int o0 = Poff[0 * TP + p], o1 = Poff[1 * TP + p];
                const int o2 = Poff[2 * TP + p], o3 = Poff[3 * TP + p];
                const float f0 = Pwt[0 * TP + p], f1 = Pwt[1 * TP + p];
                const float f2 = Pwt[2 * TP + p], f3 = Pwt[3 * TP + p];
                const float* xb = x + (size_t)(n * CIN + cb * KC + c0) * HWc;
                v8bf pk;
#pragma unroll
                for (int j = 0; j < 8; ++j) {
                    const float* xc = xb + (size_t)j * HWc;
                    const float v = f0 * xc[o0] + f1 * xc[o1] + f2 * xc[o2] + f3 * xc[o3];
                    pk[j] = f2bf(v);
                }
                *(v8bf*)(&Vlds[p * VSTR + c0]) = pk;
            }
            __syncthreads();

            // ---- WMMA compute: 2 (Cout) x 4 (pixel) tiles per wave ----
            v16bf A[2], B[4];
#pragma unroll
            for (int t = 0; t < 2; ++t) {
                // A 16x32 bf16 layout: lane half selects K groups [8h,8h+8) and [16+8h,24+8h)
                const int orow = wave * 32 + t * 16 + m16;
                v8bf alo = *(const v8bf*)(&Wlds[orow * WSTR + half * 8]);
                v8bf ahi = *(const v8bf*)(&Wlds[orow * WSTR + 16 + half * 8]);
                A[t] = __builtin_shufflevector(alo, ahi, 0,1,2,3,4,5,6,7,8,9,10,11,12,13,14,15);
            }
#pragma unroll
            for (int u = 0; u < 4; ++u) {
                // B 32x16 bf16 layout: lane col = m16, lane half selects K groups [8h,8h+8) and [16+8h,24+8h)
                const int prow = u * 16 + m16;
                v8bf blo = *(const v8bf*)(&Vlds[prow * VSTR + half * 8]);
                v8bf bhi = *(const v8bf*)(&Vlds[prow * VSTR + 16 + half * 8]);
                B[u] = __builtin_shufflevector(blo, bhi, 0,1,2,3,4,5,6,7,8,9,10,11,12,13,14,15);
            }
#pragma unroll
            for (int t = 0; t < 2; ++t)
#pragma unroll
                for (int u = 0; u < 4; ++u)
                    acc[t][u] = __builtin_amdgcn_wmma_f32_16x16x32_bf16(
                        /*neg_a=*/false, A[t], /*neg_b=*/false, B[u],
                        /*c_mod=*/(short)0, acc[t][u],
                        /*reuse_a=*/false, /*reuse_b=*/false);
            __syncthreads();
        }
    }

    // ---- epilogue: D layout (VGPR r: M = r + 8*half, N = m16), add bias ----
#pragma unroll
    for (int t = 0; t < 2; ++t) {
#pragma unroll
        for (int u = 0; u < 4; ++u) {
#pragma unroll
            for (int r = 0; r < 8; ++r) {
                const int o  = wave * 32 + t * 16 + half * 8 + r;
                const int hw = hw0 + u * 16 + m16;
                out[(size_t)(n * COUT + o) * HWc + hw] = acc[t][u][r] + bias[o];
            }
        }
    }
}

extern "C" void kernel_launch(void* const* d_in, const int* in_sizes, int n_in,
                              void* d_out, int out_size, void* d_ws, size_t ws_size,
                              hipStream_t stream) {
    (void)in_sizes; (void)n_in; (void)out_size; (void)d_ws; (void)ws_size;
    const float* x    = (const float*)d_in[0];
    const float* offs = (const float*)d_in[1];
    const float* msk  = (const float*)d_in[2];
    const float* wgt  = (const float*)d_in[3];
    const float* bias = (const float*)d_in[4];
    float* out = (float*)d_out;

    const int blocks = Nb * (HWc / TP);   // 8 * 49 = 392 workgroups
    mdcn_wmma_kernel<<<blocks, 256, 0, stream>>>(x, offs, msk, wgt, bias, out);
}
